// MultiHeadAttention_3324304687174
// MI455X (gfx1250) — matrix operand order repaired and checked by reading, not compiled, not run on hardware
//
#include <hip/hip_runtime.h>
#include <hip/hip_bf16.h>

// MHA: B=4, S=2048, D=1024, H=16, Dh=64.
// k1: qkv[8192,3072](bf16, in d_ws) = x @ W + b via v_wmma_f32_16x16x32_bf16.
// k2: flash attention; K/Q tiles fetched with the Tensor Data Mover
//     (tensor_load_to_lds, TENSORcnt), V staged with global_load_async_to_lds
//     (ASYNCcnt), all matmuls bf16 WMMA with f32 accumulation.

typedef __attribute__((ext_vector_type(16))) __bf16 v16bf;
typedef __attribute__((ext_vector_type(8)))  __bf16 v8bf;
typedef __attribute__((ext_vector_type(8)))  float  v8f;
typedef __attribute__((ext_vector_type(4)))  unsigned int u32x4;
typedef __attribute__((ext_vector_type(8)))  unsigned int u32x8;

#define NB 4
#define SEQ 2048
#define DIM 1024
#define NH 16
#define DH 64
#define D3 3072

union FragU { v16bf v; v8bf h[2]; };

// A-matrix 16x32 bf16 fragment (ISA 7.12.2): lane L<16 -> row=L, K=0..7 & 16..23;
// lane L>=16 -> row=L-16, K=8..15 & 24..31.
__device__ inline v16bf load_frag_a(const __bf16* base, int rowStride, int lane) {
    const int r  = lane & 15;
    const int kb = (lane < 16) ? 0 : 8;
    const __bf16* p = base + r * rowStride + kb;
    FragU u;
    u.h[0] = *(const v8bf*)(p);
    u.h[1] = *(const v8bf*)(p + 16);
    return u.v;
}

// B-matrix 32x16 bf16 fragment: lane owns column n = lane&15; lanes<16 hold
// K=0..7 & 16..23, lanes>=16 hold K=8..15 & 24..31. Data laid out with row
// index = n and contiguous K values per row.
__device__ inline v16bf load_frag_b(const __bf16* base, int rowStride, int lane) {
    const int n  = lane & 15;
    const int kb = (lane < 16) ? 0 : 8;
    const __bf16* p = base + n * rowStride + kb;
    FragU u;
    u.h[0] = *(const v8bf*)(p);
    u.h[1] = *(const v8bf*)(p + 16);
    return u.v;
}

__device__ inline v8f wmma_bf16(v16bf a, v16bf b, v8f c) {
    return __builtin_amdgcn_wmma_f32_16x16x32_bf16(
        false, a, false, b, (short)0, c, false, false);
}

// ---- gfx1250 async data movers -------------------------------------------

// Per-lane async copy of 16 bytes global -> LDS (GV mode), tracked by ASYNCcnt.
__device__ inline void async_copy_b128(void* lds_dst, const void* gsrc) {
    unsigned ldsOff = (unsigned)(uintptr_t)lds_dst;   // low 32 bits = LDS offset
    asm volatile("global_load_async_to_lds_b128 %0, %1, off"
                 :: "v"(ldsOff), "v"(gsrc) : "memory");
}

__device__ inline void wait_asynccnt0() {
    asm volatile("s_wait_asynccnt 0x0" ::: "memory");
}

// TDM 2D tile load (bf16 elements): builds D# groups 0/1 in SGPRs and issues
// tensor_load_to_lds (VADDR2/3 disabled -> 2D). Tracked by TENSORcnt.
__device__ inline void tdm_load_2d_bf16(unsigned lds_off, const void* gptr,
                                        unsigned tensor_d0, unsigned tensor_d1,
                                        unsigned tile_d0, unsigned tile_d1,
                                        unsigned stride_elems) {
    unsigned long long ga = (unsigned long long)(uintptr_t)gptr;
    unsigned galo = __builtin_amdgcn_readfirstlane((unsigned)ga);
    unsigned gahi = __builtin_amdgcn_readfirstlane((unsigned)(ga >> 32));
    unsigned loff = __builtin_amdgcn_readfirstlane(lds_off);
    u32x4 g0;
    g0[0] = 1u;                                   // count=1, user descriptor
    g0[1] = loff;                                 // lds_addr (bytes)
    g0[2] = galo;                                 // global_addr[31:0]
    g0[3] = (gahi & 0x01FFFFFFu) | (2u << 30);    // global_addr[56:32] | type=2
    u32x8 g1;
    g1[0] = 1u << 16;                             // data_size=1 -> 2-byte elems
    g1[1] = (tensor_d0 & 0xFFFFu) << 16;          // tensor_dim0[15:0]
    g1[2] = (tensor_d0 >> 16) | ((tensor_d1 & 0xFFFFu) << 16);
    g1[3] = (tensor_d1 >> 16) | ((tile_d0 & 0xFFFFu) << 16);
    g1[4] = tile_d1 & 0xFFFFu;                    // tile_dim1 (tile_dim2 = 0)
    g1[5] = stride_elems;                         // tensor_dim0_stride[31:0]
    g1[6] = 0u;
    g1[7] = 0u;
    asm volatile("tensor_load_to_lds %0, %1" :: "s"(g0), "s"(g1) : "memory");
}

// ---------------------------------------------------------------------------
// Kernel 1: qkv = x @ W + b   (f32 in, bf16 out)
// ---------------------------------------------------------------------------
__global__ __launch_bounds__(256) void qkv_gemm_kernel(
    const float* __restrict__ x, const float* __restrict__ W,
    const float* __restrict__ bias, __bf16* __restrict__ qkv)
{
    __shared__ __bf16 As [128][32];
    __shared__ __bf16 Bts[128][32];   // W tile transposed: [n][k]

    const int tid  = threadIdx.x;
    const int lane = tid & 31;
    const int wave = tid >> 5;
    const int wm   = wave & 3;
    const int wn   = wave >> 2;
    const int mBase = blockIdx.y * 128;
    const int nBase = blockIdx.x * 128;

    v8f acc[2][4];
#pragma unroll
    for (int i = 0; i < 2; ++i)
#pragma unroll
        for (int j = 0; j < 4; ++j) acc[i][j] = (v8f)0.0f;

    const int arow = tid >> 1;
    const int acol = (tid & 1) * 16;
    const int brow = tid >> 3;
    const int bcol = (tid & 7) * 16;

    for (int k0 = 0; k0 < DIM; k0 += 32) {
        const float4* xp = (const float4*)(x + (size_t)(mBase + arow) * DIM + k0 + acol);
        float4 f0 = xp[0], f1 = xp[1], f2 = xp[2], f3 = xp[3];
        v8bf pa0 = {(__bf16)f0.x,(__bf16)f0.y,(__bf16)f0.z,(__bf16)f0.w,
                    (__bf16)f1.x,(__bf16)f1.y,(__bf16)f1.z,(__bf16)f1.w};
        v8bf pa1 = {(__bf16)f2.x,(__bf16)f2.y,(__bf16)f2.z,(__bf16)f2.w,
                    (__bf16)f3.x,(__bf16)f3.y,(__bf16)f3.z,(__bf16)f3.w};
        *(v8bf*)(&As[arow][acol])     = pa0;
        *(v8bf*)(&As[arow][acol + 8]) = pa1;

        const float4* wp = (const float4*)(W + (size_t)(k0 + brow) * D3 + nBase + bcol);
        float4 g0 = wp[0], g1 = wp[1], g2 = wp[2], g3 = wp[3];
        __bf16 tv[16] = {(__bf16)g0.x,(__bf16)g0.y,(__bf16)g0.z,(__bf16)g0.w,
                         (__bf16)g1.x,(__bf16)g1.y,(__bf16)g1.z,(__bf16)g1.w,
                         (__bf16)g2.x,(__bf16)g2.y,(__bf16)g2.z,(__bf16)g2.w,
                         (__bf16)g3.x,(__bf16)g3.y,(__bf16)g3.z,(__bf16)g3.w};
#pragma unroll
        for (int i = 0; i < 16; ++i) Bts[bcol + i][brow] = tv[i];

        if (k0 + 32 < DIM)
            __builtin_prefetch(x + (size_t)(mBase + arow) * DIM + k0 + 32 + acol, 0, 0);

        __syncthreads();

        v16bf afrag[2];
#pragma unroll
        for (int ti = 0; ti < 2; ++ti)
            afrag[ti] = load_frag_a(&As[wm * 32 + ti * 16][0], 32, lane);
#pragma unroll
        for (int tj = 0; tj < 4; ++tj) {
            v16bf bfrag = load_frag_b(&Bts[wn * 64 + tj * 16][0], 32, lane);
#pragma unroll
            for (int ti = 0; ti < 2; ++ti)
                acc[ti][tj] = wmma_bf16(afrag[ti], bfrag, acc[ti][tj]);
        }
        __syncthreads();
    }

    const int half = lane >> 4;
    const int nl   = lane & 15;
#pragma unroll
    for (int ti = 0; ti < 2; ++ti)
#pragma unroll
        for (int tj = 0; tj < 4; ++tj)
#pragma unroll
            for (int r = 0; r < 8; ++r) {
                const int row = mBase + wm * 32 + ti * 16 + r + half * 8;
                const int col = nBase + wn * 64 + tj * 16 + nl;
                float v = acc[ti][tj][r] + bias[col];
                qkv[(size_t)row * D3 + col] = (__bf16)v;
            }
}

// ---------------------------------------------------------------------------
// Kernel 2: flash attention over bf16 qkv (TDM + async-LDS data movement).
// ---------------------------------------------------------------------------
__global__ __launch_bounds__(256) void attn_kernel(
    const __bf16* __restrict__ qkv, float* __restrict__ out)
{
    __shared__ __bf16 Qs[128][64];     // 16 KB, TDM-filled
    __shared__ __bf16 Ks[32][64];      //  4 KB, TDM-filled per iteration
    __shared__ __bf16 Vrow[32][64];    //  4 KB, async-LDS staged V rows
    __shared__ __bf16 Vt[64][32];      //  4 KB, transposed for PV B-frags
    __shared__ __bf16 Ps[8][16][32];   //  8 KB, per-wave P patch

    const int tid  = threadIdx.x;
    const int lane = tid & 31;
    const int wave = tid >> 5;

    const int qTiles = SEQ / 128;
    const int bh = blockIdx.x / qTiles;
    const int qt = blockIdx.x % qTiles;
    const int b  = bh / NH;
    const int h  = bh % NH;

    const __bf16* qbase = qkv + (size_t)b * SEQ * D3 + h * DH;
    const __bf16* kbase = qbase + DIM;
    const __bf16* vbase = qbase + 2 * DIM;

    // --- Q tile (128x64) via Tensor Data Mover, issued once by wave 0 ---
    if (wave == 0) {
        tdm_load_2d_bf16((unsigned)(uintptr_t)&Qs[0][0],
                         qbase + (size_t)(qt * 128) * D3,
                         /*tensor_d0=*/DH, /*tensor_d1=*/SEQ,
                         /*tile_d0=*/DH, /*tile_d1=*/128,
                         /*stride=*/D3);
        __builtin_amdgcn_s_wait_tensorcnt(0);
    }
    __syncthreads();

    v16bf qfrag[2];
    qfrag[0] = load_frag_a(&Qs[wave * 16][0],  64, lane);
    qfrag[1] = load_frag_a(&Qs[wave * 16][32], 64, lane);

    float m_run[8], l_run[8];
#pragma unroll
    for (int r = 0; r < 8; ++r) { m_run[r] = -1e30f; l_run[r] = 0.0f; }
    v8f oacc[4];
#pragma unroll
    for (int t = 0; t < 4; ++t) oacc[t] = (v8f)0.0f;

    const float scale = 0.125f;
    const int half = lane >> 4;
    const int nl   = lane & 15;
    const int vr = tid >> 3;           // 0..31 kv row
    const int vc = (tid & 7) * 8;      // 0..56 d col

    for (int j0 = 0; j0 < SEQ; j0 += 32) {
        __syncthreads();   // previous iteration done reading Ks/Vt

        // K tile (32x64) via TDM; V rows via per-lane async global->LDS copies.
        if (wave == 0) {
            tdm_load_2d_bf16((unsigned)(uintptr_t)&Ks[0][0],
                             kbase + (size_t)j0 * D3,
                             DH, SEQ, DH, 32, D3);
        }
        async_copy_b128(&Vrow[vr][vc], vbase + (size_t)(j0 + vr) * D3 + vc);
        wait_asynccnt0();
        if (wave == 0) __builtin_amdgcn_s_wait_tensorcnt(0);
        __syncthreads();

        // transpose V (LDS -> LDS) for contiguous B-fragments
        {
            v8bf vv = *(const v8bf*)(&Vrow[vr][vc]);
#pragma unroll
            for (int i = 0; i < 8; ++i) Vt[vc + i][vr] = vv[i];
        }
        __syncthreads();

        // --- scores: S[16 x 32] = Q(16x64) . K^T(64x32) ---
        v8f sacc[2];
        sacc[0] = (v8f)0.0f; sacc[1] = (v8f)0.0f;
#pragma unroll
        for (int tj = 0; tj < 2; ++tj)
#pragma unroll
            for (int ds = 0; ds < 2; ++ds) {
                v16bf kf = load_frag_b(&Ks[tj * 16][ds * 32], 64, lane);
                sacc[tj] = wmma_bf16(qfrag[ds], kf, sacc[tj]);
            }

        // --- online softmax ---
#pragma unroll
        for (int r = 0; r < 8; ++r) {
            float s0 = sacc[0][r] * scale;
            float s1 = sacc[1][r] * scale;
            float mx = fmaxf(s0, s1);
#pragma unroll
            for (int m = 8; m >= 1; m >>= 1) mx = fmaxf(mx, __shfl_xor(mx, m, 32));
            float mnew = fmaxf(m_run[r], mx);
            float p0 = __expf(s0 - mnew);
            float p1 = __expf(s1 - mnew);
            float rs = p0 + p1;
#pragma unroll
            for (int m = 8; m >= 1; m >>= 1) rs += __shfl_xor(rs, m, 32);
            float corr = __expf(m_run[r] - mnew);
            l_run[r] = l_run[r] * corr + rs;
            m_run[r] = mnew;
#pragma unroll
            for (int t = 0; t < 4; ++t) oacc[t][r] *= corr;
            Ps[wave][r + half * 8][nl]      = (__bf16)p0;
            Ps[wave][r + half * 8][16 + nl] = (__bf16)p1;
        }

        // --- out += P(16x32) . V(32x64) ---
        v16bf pf = load_frag_a(&Ps[wave][0][0], 32, lane);
#pragma unroll
        for (int t = 0; t < 4; ++t) {
            v16bf vf = load_frag_b(&Vt[t * 16][0], 32, lane);
            oacc[t] = wmma_bf16(pf, vf, oacc[t]);
        }
    }

    // --- normalize, store f32 in raw [B,H,S,Dh] order (== faithful reshape) ---
#pragma unroll
    for (int t = 0; t < 4; ++t)
#pragma unroll
        for (int r = 0; r < 8; ++r) {
            const int qrow = qt * 128 + wave * 16 + r + half * 8;
            const int dcol = t * 16 + nl;
            out[((size_t)(b * NH + h) * SEQ + qrow) * DH + dcol] = oacc[t][r] / l_run[r];
        }
}

extern "C" void kernel_launch(void* const* d_in, const int* in_sizes, int n_in,
                              void* d_out, int out_size, void* d_ws, size_t ws_size,
                              hipStream_t stream) {
    const float* x  = (const float*)d_in[0];
    const float* Wq = (const float*)d_in[1];
    const float* bq = (const float*)d_in[2];
    __bf16* qkv = (__bf16*)d_ws;                 // [8192,3072] bf16 (48 MB)
    float* out  = (float*)d_out;

    dim3 g1(D3 / 128, (NB * SEQ) / 128);
    qkv_gemm_kernel<<<g1, 256, 0, stream>>>(x, Wq, bq, qkv);

    dim3 g2(NB * NH * (SEQ / 128));
    attn_kernel<<<g2, 256, 0, stream>>>(qkv, out);
}
